// RelationalMessagePassingModule_10170482557018
// MI455X (gfx1250) — compile-verified, matrix operand order checked by reading, not run on hardware
//
#include <hip/hip_runtime.h>
#include <math.h>

// ---------------------------------------------------------------------------
// RelationalMessagePassing on gfx1250: bf16 WMMA GEMMs, fused gather+MLP+
// smooth-max aggregation, f32 atomics for segment max / exp-sum.
// ---------------------------------------------------------------------------

#define NN      100000
#define EMB     128
#define SMOOTHF 12.0f
#define NLAYERS 4

typedef __attribute__((ext_vector_type(16))) __bf16 v16bf;
typedef __attribute__((ext_vector_type(8)))  __bf16 v8bf;
typedef __attribute__((ext_vector_type(4)))  __bf16 v4bf;
typedef __attribute__((ext_vector_type(8)))  float  v8f;
typedef __attribute__((ext_vector_type(4)))  float  v4f;

// Load one lane's A/B fragment (16 bf16) for v_wmma_f32_16x16x32_bf16.
// Per ISA layout: lanes 0-15 hold K[0..7],K[16..23]; lanes 16-31 hold
// K[8..15],K[24..31] (khalf = 0 or 8). Two 16B loads.
__device__ __forceinline__ v16bf rmp_frag16(const __bf16* rowbase, int k0, int khalf) {
  v8bf lo = *(const v8bf*)(rowbase + k0 + khalf);
  v8bf hi = *(const v8bf*)(rowbase + k0 + 16 + khalf);
  return __builtin_shufflevector(lo, hi, 0,1,2,3,4,5,6,7,8,9,10,11,12,13,14,15);
}

// mish(x) = x * tanh(softplus(x)).  With t = e^x:
//   tanh(ln(1+t)) = ((1+t)^2 - 1) / ((1+t)^2 + 1) = (t^2+2t)/(t^2+2t+2)
// One v_exp_f32 + FMA + fast rcp; clamp at 20 (mish(x)->x, t^2 <= 2.4e17).
__device__ __forceinline__ float rmp_mish(float x) {
  float t = __expf(fminf(x, 20.0f));
  float u = __fmaf_rn(t, t, t + t);          // t^2 + 2t
  return x * __fdividef(u, u + 2.0f);
}

// Float atomic max via signed-max / unsigned-min ordering trick.
// seg_max is initialized to -inf (0xFF800000), which orders correctly on
// both paths. -0 canonicalized to +0 first.
__device__ __forceinline__ void rmp_atomic_max_f32(float* addr, float v) {
  v += 0.0f;
  if (v >= 0.0f) atomicMax((int*)addr, __float_as_int(v));
  else           atomicMin((unsigned int*)addr, (unsigned int)__float_as_int(v));
}

// ---------------------------------------------------------------------------

__global__ __launch_bounds__(256) void rmp_copy_kernel(const float* __restrict__ a,
                                                       float* __restrict__ b, int n) {
  int i = blockIdx.x * 256 + threadIdx.x;
  if (i < n) b[i] = a[i];
}

// w is [K][Nc] row-major f32; produce wT = [Nc][K] row-major bf16 so the
// WMMA B-operand (column-major striping) becomes two contiguous 16B loads.
__global__ __launch_bounds__(256) void rmp_convT_kernel(const float* __restrict__ w,
                                                        __bf16* __restrict__ wT,
                                                        int K, int Nc) {
  int i = blockIdx.x * 256 + threadIdx.x;
  if (i < K * Nc) {
    int k = i / Nc, n = i % Nc;
    wT[n * K + k] = (__bf16)w[i];
  }
}

__global__ __launch_bounds__(256) void rmp_init_seg_kernel(float* __restrict__ smax,
                                                           float* __restrict__ ssum, int n) {
  int i = blockIdx.x * 256 + threadIdx.x;
  if (i < n) { smax[i] = -INFINITY; ssum[i] = 0.0f; }
}

__global__ __launch_bounds__(256) void rmp_fin_seg_kernel(float* __restrict__ smax, int n) {
  int i = blockIdx.x * 256 + threadIdx.x;
  if (i < n) {
    float v = smax[i];
    if (!isfinite(v)) smax[i] = 0.0f;
  }
}

// ---------------------------------------------------------------------------
// Fused relation kernel: gather -> GEMM1(mish) -> GEMM2(+residual,+bias) ->
// atomic aggregation. PASS 0: seg_max via atomic max. PASS 1: seg_sum +=
// exp(SMOOTH*(msg - seg_max)). Messages are recomputed in pass 1 instead of
// being materialized (saves 384MB and its HBM traffic).
// ---------------------------------------------------------------------------
template <int AR, int PASS>
__global__ __launch_bounds__(256) void rmp_msg_kernel(
    const float* __restrict__ h, const int* __restrict__ av,
    const __bf16* __restrict__ w_inT, const float* __restrict__ b_in,
    const __bf16* __restrict__ w_outT, const float* __restrict__ b_out,
    float* __restrict__ seg_max, float* __restrict__ seg_sum, int A)
{
  constexpr int D   = AR * EMB;   // 128 or 256
  constexpr int TM  = 32;         // atom rows per block
  constexpr int LDW = D + 8;      // bf16 row stride (528B or 272B, 16B aligned)
  constexpr int NT  = D / 16;     // 16x16 column tiles

  __shared__ __align__(16) __bf16 xs[TM * LDW];   // gathered inputs (bf16)
  __shared__ __align__(16) __bf16 hs[TM * LDW];   // hidden after mish (bf16)
  __shared__ int avs[TM * AR];

  const int tid = threadIdx.x;
  const int a0  = blockIdx.x * TM;

  for (int i = tid; i < TM * AR; i += 256) {
    int a = a0 + i / AR;
    avs[i] = (a < A) ? av[a * AR + (i % AR)] : 0;
  }
  __syncthreads();

  // Gather h rows -> LDS bf16, vectorized: 16B global loads, 8B DS stores.
  // A 4-column group never straddles the E=128 boundary, so one node per vec.
  for (int e = tid; e < TM * (D / 4); e += 256) {
    int r  = e / (D / 4);
    int c4 = (e % (D / 4)) * 4;
    v4f v = {0.0f, 0.0f, 0.0f, 0.0f};
    if (a0 + r < A) {
      int node = avs[r * AR + (c4 >> 7)];
      v = *(const v4f*)&h[node * EMB + (c4 & (EMB - 1))];
    }
    v4bf o;
    o[0] = (__bf16)v[0]; o[1] = (__bf16)v[1];
    o[2] = (__bf16)v[2]; o[3] = (__bf16)v[3];
    *(v4bf*)&xs[r * LDW + c4] = o;
  }
  __syncthreads();

  const int wave  = tid >> 5;
  const int lane  = tid & 31;
  const int l16   = lane & 15;
  const int lhalf = lane >> 4;
  const int khalf = lhalf * 8;
  const int rt    = wave & 1;   // 2 row tiles of 16
  const int grp   = wave >> 1;  // 4 column groups

  // ---- GEMM1: hid = mish(x @ w_in + b_in) -------------------------------
  for (int j = 0; j < NT / 4; ++j) {
    const int n0 = (grp + 4 * j) * 16;
    const float bias = b_in[n0 + l16];
    v8f acc;
#pragma unroll
    for (int r = 0; r < 8; ++r) acc[r] = bias;
    const __bf16* arow = &xs[(rt * 16 + l16) * LDW];
    const __bf16* brow = &w_inT[(size_t)(n0 + l16) * D];
#pragma unroll
    for (int kb = 0; kb < D / 32; ++kb) {
      v16bf af = rmp_frag16(arow, kb * 32, khalf);
      v16bf bf = rmp_frag16(brow, kb * 32, khalf);
      acc = __builtin_amdgcn_wmma_f32_16x16x32_bf16(false, af, false, bf,
                                                    (short)0, acc, false, false);
    }
#pragma unroll
    for (int r = 0; r < 8; ++r) {
      int mm = rt * 16 + r + 8 * lhalf;          // C/D layout: VGPR r, half-lane
      hs[mm * LDW + n0 + l16] = (__bf16)rmp_mish(acc[r]);
    }
  }
  __syncthreads();

  // ---- GEMM2 + residual, then aggregate ---------------------------------
  for (int j = 0; j < NT / 4; ++j) {
    const int n0 = (grp + 4 * j) * 16;
    const float bias = b_out[n0 + l16];
    v8f acc;
#pragma unroll
    for (int r = 0; r < 8; ++r) acc[r] = bias;
    const __bf16* arow = &hs[(rt * 16 + l16) * LDW];
    const __bf16* brow = &w_outT[(size_t)(n0 + l16) * D];
#pragma unroll
    for (int kb = 0; kb < D / 32; ++kb) {
      v16bf af = rmp_frag16(arow, kb * 32, khalf);
      v16bf bf = rmp_frag16(brow, kb * 32, khalf);
      acc = __builtin_amdgcn_wmma_f32_16x16x32_bf16(false, af, false, bf,
                                                    (short)0, acc, false, false);
    }
#pragma unroll
    for (int r = 0; r < 8; ++r) {
      const int mm = rt * 16 + r + 8 * lhalf;
      if (a0 + mm < A) {
        const int c    = n0 + l16;
        const float val  = acc[r] + (float)xs[mm * LDW + c];   // residual
        const int node   = avs[mm * AR + (c >> 7)];
        const int off    = node * EMB + (c & (EMB - 1));
        if (PASS == 0) {
          rmp_atomic_max_f32(&seg_max[off], val);
        } else {
          float e = __expf(SMOOTHF * (val - seg_max[off]));
          atomicAdd(&seg_sum[off], e);
        }
      }
    }
  }
}

// ---------------------------------------------------------------------------
// Update MLP: upd_in = [logsumexp, h]; h += mish(upd_in@Wi+bi)@Wo+bo.
// Row-local, so in-place on h is safe across blocks.
// ---------------------------------------------------------------------------
__global__ __launch_bounds__(256) void rmp_update_kernel(
    float* __restrict__ h,
    const float* __restrict__ seg_max, const float* __restrict__ seg_sum,
    const __bf16* __restrict__ w_inT, const float* __restrict__ b_in,
    const __bf16* __restrict__ w_outT, const float* __restrict__ b_out)
{
  constexpr int D   = 2 * EMB;  // 256
  constexpr int TM  = 32;
  constexpr int LDW = D + 8;

  __shared__ __align__(16) __bf16 xs[TM * LDW];
  __shared__ __align__(16) __bf16 hs[TM * LDW];

  const int tid   = threadIdx.x;
  const int n0row = blockIdx.x * TM;

  for (int e = tid; e < TM * D; e += 256) {
    int r = e / D, c = e % D;
    int node = n0row + r;
    float v = 0.0f;
    if (node < NN) {
      if (c < EMB) {
        float s = seg_sum[node * EMB + c] + 1e-16f;
        v = __logf(s) * (1.0f / SMOOTHF) + seg_max[node * EMB + c];
      } else {
        v = h[node * EMB + (c - EMB)];
      }
    }
    xs[r * LDW + c] = (__bf16)v;
  }
  __syncthreads();

  const int wave  = tid >> 5;
  const int lane  = tid & 31;
  const int l16   = lane & 15;
  const int lhalf = lane >> 4;
  const int khalf = lhalf * 8;
  const int rt    = wave & 1;
  const int grp   = wave >> 1;

  // GEMM1: 32x256 tile, 16 column tiles (4 per wave)
  for (int j = 0; j < 4; ++j) {
    const int n0 = (grp + 4 * j) * 16;
    const float bias = b_in[n0 + l16];
    v8f acc;
#pragma unroll
    for (int r = 0; r < 8; ++r) acc[r] = bias;
    const __bf16* arow = &xs[(rt * 16 + l16) * LDW];
    const __bf16* brow = &w_inT[(size_t)(n0 + l16) * D];
#pragma unroll
    for (int kb = 0; kb < D / 32; ++kb) {
      v16bf af = rmp_frag16(arow, kb * 32, khalf);
      v16bf bf = rmp_frag16(brow, kb * 32, khalf);
      acc = __builtin_amdgcn_wmma_f32_16x16x32_bf16(false, af, false, bf,
                                                    (short)0, acc, false, false);
    }
#pragma unroll
    for (int r = 0; r < 8; ++r) {
      int mm = rt * 16 + r + 8 * lhalf;
      hs[mm * LDW + n0 + l16] = (__bf16)rmp_mish(acc[r]);
    }
  }
  __syncthreads();

  // GEMM2: output 128 cols -> 8 tiles (2 per wave); outer residual in f32.
  for (int j = 0; j < 2; ++j) {
    const int n0 = (grp + 4 * j) * 16;
    const float bias = b_out[n0 + l16];
    v8f acc;
#pragma unroll
    for (int r = 0; r < 8; ++r) acc[r] = bias;
    const __bf16* arow = &hs[(rt * 16 + l16) * LDW];
    const __bf16* brow = &w_outT[(size_t)(n0 + l16) * D];
#pragma unroll
    for (int kb = 0; kb < D / 32; ++kb) {
      v16bf af = rmp_frag16(arow, kb * 32, khalf);
      v16bf bf = rmp_frag16(brow, kb * 32, khalf);
      acc = __builtin_amdgcn_wmma_f32_16x16x32_bf16(false, af, false, bf,
                                                    (short)0, acc, false, false);
    }
#pragma unroll
    for (int r = 0; r < 8; ++r) {
      int mm = rt * 16 + r + 8 * lhalf;
      int node = n0row + mm;
      if (node < NN) {
        int c = n0 + l16;
        h[node * EMB + c] += acc[r];   // fresh f32 residual, exclusive writer
      }
    }
  }
}

// ---------------------------------------------------------------------------

extern "C" void kernel_launch(void* const* d_in, const int* in_sizes, int n_in,
                              void* d_out, int out_size, void* d_ws, size_t ws_size,
                              hipStream_t stream) {
  (void)n_in; (void)out_size; (void)ws_size;

  const float* emb      = (const float*)d_in[0];
  const int*   av_at    = (const int*)  d_in[1];
  const float* w_in_at  = (const float*)d_in[2];
  const float* b_in_at  = (const float*)d_in[3];
  const float* w_out_at = (const float*)d_in[4];
  const float* b_out_at = (const float*)d_in[5];
  const int*   av_cl    = (const int*)  d_in[6];
  const float* w_in_cl  = (const float*)d_in[7];
  const float* b_in_cl  = (const float*)d_in[8];
  const float* w_out_cl = (const float*)d_in[9];
  const float* b_out_cl = (const float*)d_in[10];
  const int*   av_on    = (const int*)  d_in[11];
  const float* w_in_on  = (const float*)d_in[12];
  const float* b_in_on  = (const float*)d_in[13];
  const float* w_out_on = (const float*)d_in[14];
  const float* b_out_on = (const float*)d_in[15];
  const float* w_u_in   = (const float*)d_in[16];
  const float* b_u_in   = (const float*)d_in[17];
  const float* w_u_out  = (const float*)d_in[18];
  const float* b_u_out  = (const float*)d_in[19];

  const int A_at = in_sizes[1] / 2;
  const int A_cl = in_sizes[6];
  const int A_on = in_sizes[11] / 2;

  float* h = (float*)d_out;            // live node embeddings (output buffer)

  // workspace carve-out
  char* p = (char*)d_ws;
  float* seg_max = (float*)p; p += (size_t)NN * EMB * sizeof(float);
  float* seg_sum = (float*)p; p += (size_t)NN * EMB * sizeof(float);
  __bf16* at_inT  = (__bf16*)p; p += (size_t)256 * 256 * sizeof(__bf16);
  __bf16* at_outT = (__bf16*)p; p += (size_t)256 * 256 * sizeof(__bf16);
  __bf16* cl_inT  = (__bf16*)p; p += (size_t)128 * 128 * sizeof(__bf16);
  __bf16* cl_outT = (__bf16*)p; p += (size_t)128 * 128 * sizeof(__bf16);
  __bf16* on_inT  = (__bf16*)p; p += (size_t)256 * 256 * sizeof(__bf16);
  __bf16* on_outT = (__bf16*)p; p += (size_t)256 * 256 * sizeof(__bf16);
  __bf16* u_inT   = (__bf16*)p; p += (size_t)256 * 256 * sizeof(__bf16);
  __bf16* u_outT  = (__bf16*)p; p += (size_t)256 * 128 * sizeof(__bf16);

  auto cdiv = [](int a, int b) { return (a + b - 1) / b; };
  const int SEG = NN * EMB;

  // one-time (per call) bf16 transposed weight prep
  rmp_convT_kernel<<<cdiv(256 * 256, 256), 256, 0, stream>>>(w_in_at,  at_inT,  256, 256);
  rmp_convT_kernel<<<cdiv(256 * 256, 256), 256, 0, stream>>>(w_out_at, at_outT, 256, 256);
  rmp_convT_kernel<<<cdiv(128 * 128, 256), 256, 0, stream>>>(w_in_cl,  cl_inT,  128, 128);
  rmp_convT_kernel<<<cdiv(128 * 128, 256), 256, 0, stream>>>(w_out_cl, cl_outT, 128, 128);
  rmp_convT_kernel<<<cdiv(256 * 256, 256), 256, 0, stream>>>(w_in_on,  on_inT,  256, 256);
  rmp_convT_kernel<<<cdiv(256 * 256, 256), 256, 0, stream>>>(w_out_on, on_outT, 256, 256);
  rmp_convT_kernel<<<cdiv(256 * 256, 256), 256, 0, stream>>>(w_u_in,   u_inT,   256, 256);
  rmp_convT_kernel<<<cdiv(256 * 128, 256), 256, 0, stream>>>(w_u_out,  u_outT,  256, 128);

  rmp_copy_kernel<<<cdiv(SEG, 256), 256, 0, stream>>>(emb, h, SEG);

  const dim3 blk(256);
  for (int layer = 0; layer < NLAYERS; ++layer) {
    rmp_init_seg_kernel<<<cdiv(SEG, 256), blk, 0, stream>>>(seg_max, seg_sum, SEG);

    // pass 0: segment max
    rmp_msg_kernel<2, 0><<<cdiv(A_at, 32), blk, 0, stream>>>(
        h, av_at, at_inT, b_in_at, at_outT, b_out_at, seg_max, seg_sum, A_at);
    rmp_msg_kernel<1, 0><<<cdiv(A_cl, 32), blk, 0, stream>>>(
        h, av_cl, cl_inT, b_in_cl, cl_outT, b_out_cl, seg_max, seg_sum, A_cl);
    rmp_msg_kernel<2, 0><<<cdiv(A_on, 32), blk, 0, stream>>>(
        h, av_on, on_inT, b_in_on, on_outT, b_out_on, seg_max, seg_sum, A_on);

    rmp_fin_seg_kernel<<<cdiv(SEG, 256), blk, 0, stream>>>(seg_max, SEG);

    // pass 1: sum of exp(SMOOTH*(msg - seg_max))
    rmp_msg_kernel<2, 1><<<cdiv(A_at, 32), blk, 0, stream>>>(
        h, av_at, at_inT, b_in_at, at_outT, b_out_at, seg_max, seg_sum, A_at);
    rmp_msg_kernel<1, 1><<<cdiv(A_cl, 32), blk, 0, stream>>>(
        h, av_cl, cl_inT, b_in_cl, cl_outT, b_out_cl, seg_max, seg_sum, A_cl);
    rmp_msg_kernel<2, 1><<<cdiv(A_on, 32), blk, 0, stream>>>(
        h, av_on, on_inT, b_in_on, on_outT, b_out_on, seg_max, seg_sum, A_on);

    // update MLP + outer residual (in place on h)
    rmp_update_kernel<<<cdiv(NN, 32), blk, 0, stream>>>(
        h, seg_max, seg_sum, u_inT, b_u_in, u_outT, b_u_out);
  }
}
